// MultiHeadAttention_10213432230201
// MI455X (gfx1250) — matrix operand order repaired and checked by reading, not compiled, not run on hardware
//
#include <hip/hip_runtime.h>

// Problem constants (from reference): B=2, S=2048, C=1024, H=16, D=64
#define BB    2
#define SEQ   2048
#define CDIM  1024
#define HEADS 16
#define HDIM  64

typedef __attribute__((ext_vector_type(16))) __bf16 v16bf;
typedef __attribute__((ext_vector_type(8)))  __bf16 bf16x8;
typedef __attribute__((ext_vector_type(8)))  float  v8f;

__device__ __forceinline__ bf16x8 ld8(const __bf16* p) {
    return *reinterpret_cast<const bf16x8*>(p);
}
__device__ __forceinline__ v16bf mk16(bf16x8 lo, bf16x8 hi) {
    v16bf r;
#pragma unroll
    for (int i = 0; i < 8; ++i) { r[i] = lo[i]; r[i + 8] = hi[i]; }
    return r;
}
// A fragment (16x32 of a row-major matrix): two b128 per lane
__device__ __forceinline__ v16bf ldA(const __bf16* row, int kk, int half) {
    return mk16(ld8(row + kk + half * 8), ld8(row + kk + 16 + half * 8));
}
// B fragment (32x16, from B^T stored row-major [N,K]): two b128 per lane
__device__ __forceinline__ v16bf ldB(const __bf16* row, int kk, int half) {
    return mk16(ld8(row + kk + half * 8), ld8(row + kk + 16 + half * 8));
}
__device__ __forceinline__ v8f wmma_bf16(v16bf a, v16bf b, v8f c) {
    return __builtin_amdgcn_wmma_f32_16x16x32_bf16(false, a, false, b, (short)0, c,
                                                   false, false);
}

// ---------------------------------------------------------------------------
// Kernel 1: fp32 -> bf16 straight conversion (for x)
// ---------------------------------------------------------------------------
__global__ void cvt_bf16_kernel(const float* __restrict__ in, __bf16* __restrict__ out, int n) {
    int i = blockIdx.x * blockDim.x + threadIdx.x;
    if (i < n) out[i] = (__bf16)in[i];
}

// ---------------------------------------------------------------------------
// Kernel 2: fp32 [G, R, Cc] -> bf16 transposed [G, Cc, R]
// ---------------------------------------------------------------------------
__global__ void cvt_transpose_kernel(const float* __restrict__ in, __bf16* __restrict__ out,
                                     int RC, int R, int Cc, int total) {
    int i = blockIdx.x * blockDim.x + threadIdx.x;
    if (i >= total) return;
    int g   = i / RC;
    int rem = i - g * RC;
    int r   = rem / Cc;
    int c   = rem - r * Cc;
    out[(size_t)g * RC + (size_t)c * R + r] = (__bf16)in[i];
}

// ---------------------------------------------------------------------------
// Kernel 3: QKV projection. One wave -> a 32x16 tile of q,k,v for one head:
// 2 A frags (x rows) x 3 B frags (Wq/Wk/Wv) -> 6 WMMAs per K step,
// double-buffered so next-step loads are in flight during current WMMAs.
//   Q  -> [B,H,S,D]   K -> [B,H,S,D]   Vt -> [B,H,D,S]
// ---------------------------------------------------------------------------
__global__ __launch_bounds__(256) void qkv_proj_kernel(
    const __bf16* __restrict__ xb,
    const __bf16* __restrict__ wqt, const __bf16* __restrict__ wkt,
    const __bf16* __restrict__ wvt,
    __bf16* __restrict__ Q, __bf16* __restrict__ K, __bf16* __restrict__ Vt) {
    int wave = threadIdx.x >> 5;
    int lane = threadIdx.x & 31;
    int n16  = lane & 15;
    int half = lane >> 4;

    int gid = blockIdx.x * 8 + wave;           // 8192 tiles total (32x16 each)
    int dt  = gid & 3;                         // D/16
    int st  = (gid >> 2) & 63;                 // S/32
    int h   = (gid >> 8) & 15;                 // heads
    int b   = gid >> 12;                       // batch

    const __bf16* Arow0 = xb + ((size_t)(b * SEQ + st * 32 + n16)) * CDIM;
    const __bf16* Arow1 = xb + ((size_t)(b * SEQ + st * 32 + 16 + n16)) * CDIM;
    const __bf16* Bq = wqt + ((size_t)(h * HDIM + dt * 16 + n16)) * CDIM;
    const __bf16* Bk = wkt + ((size_t)(h * HDIM + dt * 16 + n16)) * CDIM;
    const __bf16* Bv = wvt + ((size_t)(h * HDIM + dt * 16 + n16)) * CDIM;

    v8f cq0 = {}, ck0 = {}, cv0 = {};
    v8f cq1 = {}, ck1 = {}, cv1 = {};

    // software pipeline: fragments for step kk loaded one step ahead
    v16bf a0 = ldA(Arow0, 0, half);
    v16bf a1 = ldA(Arow1, 0, half);
    v16bf bq = ldB(Bq, 0, half);
    v16bf bk = ldB(Bk, 0, half);
    v16bf bv = ldB(Bv, 0, half);
#pragma unroll 4
    for (int kk = 32; kk < CDIM; kk += 32) {
        v16bf a0n = ldA(Arow0, kk, half);
        v16bf a1n = ldA(Arow1, kk, half);
        v16bf bqn = ldB(Bq, kk, half);
        v16bf bkn = ldB(Bk, kk, half);
        v16bf bvn = ldB(Bv, kk, half);
        cq0 = wmma_bf16(a0, bq, cq0);
        cq1 = wmma_bf16(a1, bq, cq1);
        ck0 = wmma_bf16(a0, bk, ck0);
        ck1 = wmma_bf16(a1, bk, ck1);
        cv0 = wmma_bf16(a0, bv, cv0);
        cv1 = wmma_bf16(a1, bv, cv1);
        a0 = a0n; a1 = a1n; bq = bqn; bk = bkn; bv = bvn;
    }
    cq0 = wmma_bf16(a0, bq, cq0);
    cq1 = wmma_bf16(a1, bq, cq1);
    ck0 = wmma_bf16(a0, bk, ck0);
    ck1 = wmma_bf16(a1, bk, ck1);
    cv0 = wmma_bf16(a0, bv, cv0);
    cv1 = wmma_bf16(a1, bv, cv1);

    size_t base = ((size_t)(b * HEADS + h)) * SEQ * HDIM;
    int ncol = dt * 16 + n16;
#pragma unroll
    for (int r = 0; r < 8; ++r) {
        int mrow0 = st * 32 + half * 8 + r;       // C-layout: VGPR r, halves -> M
        int mrow1 = mrow0 + 16;
        Q[base + (size_t)mrow0 * HDIM + ncol] = (__bf16)cq0[r];
        Q[base + (size_t)mrow1 * HDIM + ncol] = (__bf16)cq1[r];
        K[base + (size_t)mrow0 * HDIM + ncol] = (__bf16)ck0[r];
        K[base + (size_t)mrow1 * HDIM + ncol] = (__bf16)ck1[r];
        Vt[base + (size_t)ncol * SEQ + mrow0] = (__bf16)cv0[r];  // transposed
        Vt[base + (size_t)ncol * SEQ + mrow1] = (__bf16)cv1[r];
    }
}

// ---------------------------------------------------------------------------
// Kernel 4: causal flash attention. 4 waves/block; each wave owns 16 query
// rows, streams key/value chunks of 32 positions; V fragments are loaded
// before the softmax VALU section so they overlap it.
// ---------------------------------------------------------------------------
__global__ __launch_bounds__(128) void flash_attn_kernel(
    const __bf16* __restrict__ Q, const __bf16* __restrict__ K,
    const __bf16* __restrict__ Vt, __bf16* __restrict__ Obf) {
    __shared__ __align__(16) __bf16 plds[4][16][32];

    int wave = threadIdx.x >> 5;
    int lane = threadIdx.x & 31;
    int n16  = lane & 15;
    int half = lane >> 4;

    int blk = blockIdx.x;               // B * H * (S/64)
    int qb  = (blk & 31) * 64;          // S/64 = 32
    int h   = (blk >> 5) & 15;
    int b   = blk >> 9;
    int qw  = qb + wave * 16;           // this wave's query-row base

    size_t base = ((size_t)(b * HEADS + h)) * SEQ * HDIM;
    const __bf16* Qp = Q + base;
    const __bf16* Kp = K + base;
    const __bf16* Vp = Vt + base;

    // Q fragments for 16 rows x 64 (K split 0..31 / 32..63)
    const __bf16* qr = Qp + (size_t)(qw + n16) * HDIM;
    v16bf a0 = mk16(ld8(qr + half * 8),      ld8(qr + 16 + half * 8));
    v16bf a1 = mk16(ld8(qr + 32 + half * 8), ld8(qr + 48 + half * 8));

    float mst[8], lst[8];
    v8f ov0 = {}, ov1 = {}, ov2 = {}, ov3 = {};
#pragma unroll
    for (int r = 0; r < 8; ++r) { mst[r] = -1e30f; lst[r] = 0.0f; }

    const float sc = 0.125f * 1.44269504089f;   // D^-0.5 * log2(e): exp2 domain

    for (int tb = 0; tb < qw + 16; tb += 32) {
        // --- scores: two 16-wide key subtiles ---
        v8f s0 = {}, s1 = {};
        {
            const __bf16* kr = Kp + (size_t)(tb + n16) * HDIM;
            v16bf b0 = mk16(ld8(kr + half * 8), ld8(kr + 16 + half * 8));
            v16bf b1 = mk16(ld8(kr + 32 + half * 8), ld8(kr + 48 + half * 8));
            s0 = wmma_bf16(a0, b0, s0);
            s0 = wmma_bf16(a1, b1, s0);
        }
        {
            const __bf16* kr = Kp + (size_t)(tb + 16 + n16) * HDIM;
            v16bf b0 = mk16(ld8(kr + half * 8), ld8(kr + 16 + half * 8));
            v16bf b1 = mk16(ld8(kr + 32 + half * 8), ld8(kr + 48 + half * 8));
            s1 = wmma_bf16(a0, b0, s1);
            s1 = wmma_bf16(a1, b1, s1);
        }

        // --- V fragments for this chunk: issue now, consume after softmax ---
        const __bf16* vr0 = Vp + (size_t)(n16) * SEQ + tb + half * 8;
        const __bf16* vr1 = Vp + (size_t)(16 + n16) * SEQ + tb + half * 8;
        const __bf16* vr2 = Vp + (size_t)(32 + n16) * SEQ + tb + half * 8;
        const __bf16* vr3 = Vp + (size_t)(48 + n16) * SEQ + tb + half * 8;
        v16bf vb0 = mk16(ld8(vr0), ld8(vr0 + 16));
        v16bf vb1 = mk16(ld8(vr1), ld8(vr1 + 16));
        v16bf vb2 = mk16(ld8(vr2), ld8(vr2 + 16));
        v16bf vb3 = mk16(ld8(vr3), ld8(vr3 + 16));

        // --- scale + causal mask + chunk row-max ---
        float cm[8];
#pragma unroll
        for (int r = 0; r < 8; ++r) {
            int mrow = qw + half * 8 + r;
            float v0 = (tb + n16 <= mrow)      ? s0[r] * sc : -1e30f;
            float v1 = (tb + 16 + n16 <= mrow) ? s1[r] * sc : -1e30f;
            s0[r] = v0; s1[r] = v1;
            cm[r] = fmaxf(v0, v1);
        }
#pragma unroll
        for (int r = 0; r < 8; ++r)
#pragma unroll
            for (int m = 1; m < 16; m <<= 1)
                cm[r] = fmaxf(cm[r], __shfl_xor(cm[r], m, 32));

        // --- online softmax update ---
        float f[8], ps[8];
#pragma unroll
        for (int r = 0; r < 8; ++r) {
            float nm = fmaxf(mst[r], cm[r]);
            f[r] = exp2f(mst[r] - nm);
            mst[r] = nm;
            float p0 = exp2f(s0[r] - nm);
            float p1 = exp2f(s1[r] - nm);
            s0[r] = p0; s1[r] = p1;
            ps[r] = p0 + p1;
        }
#pragma unroll
        for (int r = 0; r < 8; ++r)
#pragma unroll
            for (int m = 1; m < 16; m <<= 1)
                ps[r] += __shfl_xor(ps[r], m, 32);
#pragma unroll
        for (int r = 0; r < 8; ++r) {
            lst[r] = lst[r] * f[r] + ps[r];
            ov0[r] *= f[r]; ov1[r] *= f[r]; ov2[r] *= f[r]; ov3[r] *= f[r];
        }

        // --- stage P (C-layout) into LDS, reload as A fragment (same-wave
        //     LDS is in-order; backend inserts the dscnt waits) ---
#pragma unroll
        for (int r = 0; r < 8; ++r) {
            int rl = half * 8 + r;
            plds[wave][rl][n16]      = (__bf16)s0[r];
            plds[wave][rl][16 + n16] = (__bf16)s1[r];
        }
        const __bf16* pr = &plds[wave][n16][0];
        v16bf pf = mk16(*(const bf16x8*)(pr + half * 8),
                        *(const bf16x8*)(pr + 16 + half * 8));

        // --- O += P * V over 4 d-tiles ---
        ov0 = wmma_bf16(pf, vb0, ov0);
        ov1 = wmma_bf16(pf, vb1, ov1);
        ov2 = wmma_bf16(pf, vb2, ov2);
        ov3 = wmma_bf16(pf, vb3, ov3);
    }

    // --- normalize and store concat-head output bf16 [B, S, H*D] ---
#pragma unroll
    for (int r = 0; r < 8; ++r) {
        float inv = 1.0f / lst[r];
        int mrow = qw + half * 8 + r;
        size_t ob = ((size_t)(b * SEQ + mrow)) * (HEADS * HDIM) + h * HDIM;
        Obf[ob + 0 * 16 + n16] = (__bf16)(ov0[r] * inv);
        Obf[ob + 1 * 16 + n16] = (__bf16)(ov1[r] * inv);
        Obf[ob + 2 * 16 + n16] = (__bf16)(ov2[r] * inv);
        Obf[ob + 3 * 16 + n16] = (__bf16)(ov3[r] * inv);
    }
}

// ---------------------------------------------------------------------------
// Kernel 5: output projection Y = Obf[B*S,1024] x Wo[1024,1024] + bo.
// 32x32 register tile per wave (2 A frags x 2 B frags -> 4 WMMAs/step),
// double-buffered K-loop.
// ---------------------------------------------------------------------------
__global__ __launch_bounds__(256) void out_proj_kernel(
    const __bf16* __restrict__ Obf, const __bf16* __restrict__ wot,
    const float* __restrict__ bo, float* __restrict__ Y) {
    int wave = threadIdx.x >> 5;
    int lane = threadIdx.x & 31;
    int n16  = lane & 15;
    int half = lane >> 4;

    int gid = blockIdx.x * 8 + wave;     // 128 * 32 = 4096 tiles of 32x32
    int nt  = gid & 31;                  // N/32
    int mt  = gid >> 5;                  // (B*S)/32

    const __bf16* Arow0 = Obf + ((size_t)(mt * 32 + n16)) * CDIM;
    const __bf16* Arow1 = Obf + ((size_t)(mt * 32 + 16 + n16)) * CDIM;
    const __bf16* Brow0 = wot + ((size_t)(nt * 32 + n16)) * CDIM;
    const __bf16* Brow1 = wot + ((size_t)(nt * 32 + 16 + n16)) * CDIM;

    v8f acc00 = {}, acc01 = {}, acc10 = {}, acc11 = {};

    v16bf a0 = ldA(Arow0, 0, half);
    v16bf a1 = ldA(Arow1, 0, half);
    v16bf b0 = ldB(Brow0, 0, half);
    v16bf b1 = ldB(Brow1, 0, half);
#pragma unroll 4
    for (int kk = 32; kk < CDIM; kk += 32) {
        v16bf a0n = ldA(Arow0, kk, half);
        v16bf a1n = ldA(Arow1, kk, half);
        v16bf b0n = ldB(Brow0, kk, half);
        v16bf b1n = ldB(Brow1, kk, half);
        acc00 = wmma_bf16(a0, b0, acc00);
        acc01 = wmma_bf16(a0, b1, acc01);
        acc10 = wmma_bf16(a1, b0, acc10);
        acc11 = wmma_bf16(a1, b1, acc11);
        a0 = a0n; a1 = a1n; b0 = b0n; b1 = b1n;
    }
    acc00 = wmma_bf16(a0, b0, acc00);
    acc01 = wmma_bf16(a0, b1, acc01);
    acc10 = wmma_bf16(a1, b0, acc10);
    acc11 = wmma_bf16(a1, b1, acc11);

    float bias0 = bo[nt * 32 + n16];
    float bias1 = bo[nt * 32 + 16 + n16];
#pragma unroll
    for (int r = 0; r < 8; ++r) {
        int m0 = mt * 32 + half * 8 + r;
        int m1 = m0 + 16;
        size_t c0 = nt * 32 + n16;
        Y[(size_t)m0 * CDIM + c0]      = acc00[r] + bias0;
        Y[(size_t)m0 * CDIM + c0 + 16] = acc01[r] + bias1;
        Y[(size_t)m1 * CDIM + c0]      = acc10[r] + bias0;
        Y[(size_t)m1 * CDIM + c0 + 16] = acc11[r] + bias1;
    }
}

// ---------------------------------------------------------------------------
// Host launcher
// ---------------------------------------------------------------------------
extern "C" void kernel_launch(void* const* d_in, const int* in_sizes, int n_in,
                              void* d_out, int out_size, void* d_ws, size_t ws_size,
                              hipStream_t stream) {
    (void)in_sizes; (void)n_in; (void)out_size; (void)ws_size;

    const float* x  = (const float*)d_in[0];
    const float* Wq = (const float*)d_in[1];
    const float* Wk = (const float*)d_in[2];
    const float* Wv = (const float*)d_in[3];
    const float* Wo = (const float*)d_in[4];
    const float* bo = (const float*)d_in[5];
    float* Y = (float*)d_out;

    char* ws = (char*)d_ws;
    const size_t MB = 1024 * 1024;
    __bf16* xb  = (__bf16*)(ws + 0);          //  8 MiB: x bf16 [B,S,C]
    __bf16* wqt = (__bf16*)(ws + 8 * MB);     //  2 MiB: Wq^T [H,D,C]
    __bf16* wkt = (__bf16*)(ws + 10 * MB);    //  2 MiB
    __bf16* wvt = (__bf16*)(ws + 12 * MB);    //  2 MiB
    __bf16* wot = (__bf16*)(ws + 14 * MB);    //  2 MiB: Wo^T [C,HD]
    __bf16* Qb  = (__bf16*)(ws + 16 * MB);    //  8 MiB: [B,H,S,D]
    __bf16* Kb  = (__bf16*)(ws + 24 * MB);    //  8 MiB: [B,H,S,D]
    __bf16* Vtb = (__bf16*)(ws + 32 * MB);    //  8 MiB: [B,H,D,S]
    __bf16* Ob  = (__bf16*)(ws + 40 * MB);    //  8 MiB: [B,S,H*D]
    // total 48 MiB of workspace

    int nX = BB * SEQ * CDIM;
    cvt_bf16_kernel<<<(nX + 255) / 256, 256, 0, stream>>>(x, xb, nX);

    int nW = HEADS * CDIM * HDIM;
    cvt_transpose_kernel<<<(nW + 255) / 256, 256, 0, stream>>>(Wq, wqt, CDIM * HDIM, CDIM, HDIM, nW);
    cvt_transpose_kernel<<<(nW + 255) / 256, 256, 0, stream>>>(Wk, wkt, CDIM * HDIM, CDIM, HDIM, nW);
    cvt_transpose_kernel<<<(nW + 255) / 256, 256, 0, stream>>>(Wv, wvt, CDIM * HDIM, CDIM, HDIM, nW);
    int nWo = CDIM * CDIM;
    cvt_transpose_kernel<<<(nWo + 255) / 256, 256, 0, stream>>>(Wo, wot, CDIM * CDIM, CDIM, CDIM, nWo);

    // 8192 wave-tiles (32x16) / 8 waves per block
    qkv_proj_kernel<<<1024, 256, 0, stream>>>(xb, wqt, wkt, wvt, Qb, Kb, Vtb);

    // B * H * (S/64) blocks of 4 waves
    flash_attn_kernel<<<BB * HEADS * (SEQ / 64), 128, 0, stream>>>(Qb, Kb, Vtb, Ob);

    // 4096 wave-tiles (32x32) / 8 waves per block
    out_proj_kernel<<<512, 256, 0, stream>>>(Ob, wot, bo, Y);
}
